// GPT_137438954040
// MI455X (gfx1250) — compile-verified, matrix operand order checked by reading, not run on hardware
//
#include <hip/hip_runtime.h>
#include <math.h>

// ---------------- model constants ----------------
#define V_SZ   32000
#define DIM    768
#define NL     12
#define DST    16
#define DCV    4
#define DIN    1536
#define HID    3072
#define BV     2048
#define BD     64
#define BSZ    2
#define SEQ    1024
#define TOK    (BSZ*SEQ)          // 2048
#define NSKIP  (NL/2)
#define SOFTCAP 30.0f
#define EPSF   1.1920929e-07f

// ---------------- WMMA types ----------------
typedef __attribute__((ext_vector_type(16))) __bf16 v16bf;
typedef __attribute__((ext_vector_type(8)))  float  v8f;

union FragBF { uint4 q[2]; v16bf v; };
union U2U    { unsigned short u[4]; uint2 v; };

__device__ __forceinline__ unsigned short f2bf(float f) {
    unsigned u = __float_as_uint(f);
    unsigned r = u + 0x7FFFu + ((u >> 16) & 1u);   // round-to-nearest-even
    return (unsigned short)(r >> 16);
}
__device__ __forceinline__ float sigm(float x) { return 1.f / (1.f + __expf(-x)); }

#define LSTR 56   // LDS row stride in halfwords (112B: 16B aligned, conflict-free)

// ---- async memory->LDS copy of one 128x32 bf16 A tile (8KB) ----
// Uses CDNA5 GLOBAL_LOAD_ASYNC_TO_LDS_B128 (ASYNCcnt-tracked, no VGPR staging).
__device__ __forceinline__ void stage_tileA_async(const unsigned short* __restrict__ A,
                                                  unsigned short* la,
                                                  int tid, int m0, int K, int k0)
{
    #pragma unroll
    for (int it = 0; it < 2; ++it) {
        int idx = tid + it * 256;
        int r = idx >> 2, sg = idx & 3;
        const unsigned short* g = A + (size_t)(m0 + r) * K + k0 + sg * 8;
        unsigned long long ga = (unsigned long long)(const void*)g;
        // low 32 bits of the generic shared-pointer == LDS byte offset
        unsigned ldso = (unsigned)(unsigned long long)(const void*)(la + r * LSTR + sg * 8);
        asm volatile("global_load_async_to_lds_b128 %0, %1, off"
                     :: "v"(ldso), "v"(ga) : "memory");
    }
}

// ---- synchronous f32 weight tile -> bf16 LDS tile (needs in-register convert) ----
__device__ __forceinline__ void stage_tileB(const float* __restrict__ W,
                                            unsigned short* lb,
                                            int tid, int n0, int N, int K, int k0)
{
    #pragma unroll
    for (int it = 0; it < 4; ++it) {
        int idx = tid + it * 256;
        int r = idx >> 3, sg = idx & 7;
        U2U o;
        if (n0 + r < N) {
            const float4 wv = *reinterpret_cast<const float4*>(
                W + (size_t)(n0 + r) * K + k0 + sg * 4);
            o.u[0] = f2bf(wv.x); o.u[1] = f2bf(wv.y);
            o.u[2] = f2bf(wv.z); o.u[3] = f2bf(wv.w);
        } else { o.u[0] = o.u[1] = o.u[2] = o.u[3] = 0; }
        *reinterpret_cast<uint2*>(lb + r * LSTR + sg * 4) = o.v;
    }
}

// ============================================================
// Generic bf16 WMMA GEMM:  C[M,N] = A[M,K](bf16) * W[N,K](f32)^T
// Double-buffered: async A-copy + W-convert for tile kt+1 overlap
// the WMMAs of tile kt.
// EPI 0: Cf = acc                     (fp32 store)
// EPI 1: Cb = bf16(relu(acc)^2)       (bf16 store, feeds next GEMM)
// EPI 2: Cf = resid + scale[n]*acc    (fused residual, fp32)
// EPI 3: Cf = SOFTCAP*tanh(acc/SOFTCAP)
// ============================================================
template<int EPI>
__global__ __launch_bounds__(256)
void gemm_bf16_wmma(const unsigned short* __restrict__ A,
                    const float*          __restrict__ W,
                    float*                __restrict__ Cf,
                    unsigned short*       __restrict__ Cb,
                    const float*          __restrict__ resid,
                    const float*          __restrict__ scale,
                    int M, int N, int K)
{
    __shared__ alignas(16) unsigned short lA[2][128 * LSTR];
    __shared__ alignas(16) unsigned short lB[2][128 * LSTR];

    const int tid  = threadIdx.x;
    const int lane = tid & 31;
    const int wave = tid >> 5;          // 0..7
    const int wm   = wave >> 2;         // 0..1 -> 64 rows each
    const int wn   = wave & 3;          // 0..3 -> 32 cols each
    const int l16  = lane & 15;
    const int h    = lane >> 4;         // lane half
    const int m0   = blockIdx.y * 128;
    const int n0   = blockIdx.x * 128;

    v8f zero;
    #pragma unroll
    for (int e = 0; e < 8; ++e) zero[e] = 0.f;
    v8f acc[4][2];
    #pragma unroll
    for (int mi = 0; mi < 4; ++mi)
        #pragma unroll
        for (int ni = 0; ni < 2; ++ni) acc[mi][ni] = zero;

    const int KT = K >> 5;

    // ---- prologue: stage tile 0 into buffer 0 ----
    stage_tileA_async(A, &lA[0][0], tid, m0, K, 0);
    stage_tileB(W, &lB[0][0], tid, n0, N, K, 0);
    asm volatile("s_wait_asynccnt 0x0" ::: "memory");
    __syncthreads();

    for (int kt = 0; kt < KT; ++kt) {
        const int cur = kt & 1;
        const int nxt = cur ^ 1;

        // ---- issue next tile while computing this one ----
        if (kt + 1 < KT) {
            const int k1 = (kt + 1) << 5;
            stage_tileA_async(A, &lA[nxt][0], tid, m0, K, k1);
            stage_tileB(W, &lB[nxt][0], tid, n0, N, K, k1);
        }
        // keep L2 ahead on the weight stream
        if (kt + 2 < KT) {
            int r = tid >> 1;
            if (n0 + r < N)
                __builtin_prefetch(W + (size_t)(n0 + r) * K + ((kt + 2) << 5), 0, 3);
        }

        // ---- fragments + WMMA on current buffer ----
        FragBF af[4], bfr[2];
        #pragma unroll
        for (int mi = 0; mi < 4; ++mi) {
            int row = wm * 64 + mi * 16 + l16;
            int off = row * LSTR + 8 * h;
            af[mi].q[0] = *reinterpret_cast<const uint4*>(&lA[cur][off]);
            af[mi].q[1] = *reinterpret_cast<const uint4*>(&lA[cur][off + 16]);
        }
        #pragma unroll
        for (int ni = 0; ni < 2; ++ni) {
            int row = wn * 32 + ni * 16 + l16;
            int off = row * LSTR + 8 * h;
            bfr[ni].q[0] = *reinterpret_cast<const uint4*>(&lB[cur][off]);
            bfr[ni].q[1] = *reinterpret_cast<const uint4*>(&lB[cur][off + 16]);
        }
        #pragma unroll
        for (int mi = 0; mi < 4; ++mi)
            #pragma unroll
            for (int ni = 0; ni < 2; ++ni)
                acc[mi][ni] = __builtin_amdgcn_wmma_f32_16x16x32_bf16(
                    false, af[mi].v, false, bfr[ni].v,
                    (short)0, acc[mi][ni], false, false);

        // next tile's async LDS writes must be visible to all before reuse
        asm volatile("s_wait_asynccnt 0x0" ::: "memory");
        __syncthreads();
    }

    // ---- epilogue ----
    #pragma unroll
    for (int mi = 0; mi < 4; ++mi) {
        #pragma unroll
        for (int ni = 0; ni < 2; ++ni) {
            int n = n0 + wn * 32 + ni * 16 + l16;
            if (n < N) {
                #pragma unroll
                for (int r = 0; r < 8; ++r) {
                    int m = m0 + wm * 64 + mi * 16 + 8 * h + r;
                    float v = acc[mi][ni][r];
                    size_t o = (size_t)m * N + n;
                    if (EPI == 0) Cf[o] = v;
                    if (EPI == 1) { float t = fmaxf(v, 0.f); Cb[o] = f2bf(t * t); }
                    if (EPI == 2) Cf[o] = resid[o] + scale[n] * v;
                    if (EPI == 3) Cf[o] = SOFTCAP * tanhf(v * (1.0f / SOFTCAP));
                }
            }
        }
    }
}

// ============================================================
// Embedding: x0 = rmsnorm(tok_emb[t] + bs * big_proj @ big_emb[hash])
// one block per token
// ============================================================
__global__ __launch_bounds__(256)
void embed_kernel(const int* __restrict__ ids, const float* __restrict__ tok_emb,
                  const float* __restrict__ big_emb, const float* __restrict__ big_proj,
                  const float* __restrict__ big_scale, float* __restrict__ xout)
{
    __shared__ float red[256];
    __shared__ float be[BD];
    const int tok = blockIdx.x;
    const int s   = tok & (SEQ - 1);
    const int tid = threadIdx.x;
    int t = ids[tok];
    int hidx;
    if (s == 0) hidx = BV - 1;
    else {
        int tp = ids[tok - 1];
        hidx = ((36313 * t) ^ (27191 * tp)) % (BV - 1);
    }
    if (tid < BD) be[tid] = big_emb[(size_t)hidx * BD + tid];
    __syncthreads();
    float bs = big_scale[0];
    float v[3]; float ss = 0.f;
    #pragma unroll
    for (int j = 0; j < 3; ++j) {
        int d = tid + j * 256;
        const float* bp = big_proj + (size_t)d * BD;
        float acc = 0.f;
        #pragma unroll
        for (int k = 0; k < BD; ++k) acc += bp[k] * be[k];
        float e = tok_emb[(size_t)t * DIM + d] + bs * acc;
        v[j] = e; ss += e * e;
    }
    red[tid] = ss; __syncthreads();
    for (int st = 128; st > 0; st >>= 1) { if (tid < st) red[tid] += red[tid + st]; __syncthreads(); }
    float rms = rsqrtf(red[0] / (float)DIM + EPSF);
    #pragma unroll
    for (int j = 0; j < 3; ++j) {
        int d = tid + j * 256;
        xout[(size_t)tok * DIM + d] = v[j] * rms;
    }
}

// ============================================================
// Optional skip-add + smear gate + rmsnorm -> bf16 operand
// do_smear=1: xs_out gets the smeared residual stream (fp32)
// ============================================================
__global__ __launch_bounds__(256)
void smear_norm_kernel(const float* __restrict__ xin,
                       const float* __restrict__ skip,
                       const float* __restrict__ skw,
                       const float* __restrict__ gate,
                       float* __restrict__ xs_out,
                       unsigned short* __restrict__ xn_out,
                       int do_smear)
{
    __shared__ float red[256];
    const int tok = blockIdx.x;
    const int s   = tok & (SEQ - 1);
    const int tid = threadIdx.x;
    float v[3]; float ss = 0.f;
    #pragma unroll
    for (int j = 0; j < 3; ++j) {
        int d = tid + j * 256;
        size_t o = (size_t)tok * DIM + d;
        float cur = xin[o];
        if (skip) cur += skw[d] * skip[o];
        if (do_smear) {
            float prev = 0.f;
            if (s > 0) {
                size_t op = o - DIM;
                prev = xin[op];
                if (skip) prev += skw[d] * skip[op];
            }
            float g = sigm(gate[d]);
            cur = (1.f - g) * cur + g * prev;
            xs_out[o] = cur;
        }
        v[j] = cur; ss += cur * cur;
    }
    red[tid] = ss; __syncthreads();
    for (int st = 128; st > 0; st >>= 1) { if (tid < st) red[tid] += red[tid + st]; __syncthreads(); }
    float rms = rsqrtf(red[0] / (float)DIM + EPSF);
    #pragma unroll
    for (int j = 0; j < 3; ++j) {
        int d = tid + j * 256;
        xn_out[(size_t)tok * DIM + d] = f2bf(v[j] * rms);
    }
}

// ============================================================
// Causal depthwise conv (DCV=4) + bias + SiLU ; writes fp32 + bf16
// ============================================================
__global__ __launch_bounds__(256)
void conv_kernel(const float* __restrict__ xz, const float* __restrict__ cw,
                 const float* __restrict__ cb, float* __restrict__ xc,
                 unsigned short* __restrict__ xcb)
{
    int idx = blockIdx.x * 256 + threadIdx.x;    // [0, TOK*DIN)
    int tok = idx / DIN;
    int d   = idx - tok * DIN;
    int s   = tok & (SEQ - 1);
    const float* w = cw + (size_t)d * DCV;
    float acc = cb[d];
    #pragma unroll
    for (int k = 0; k < DCV; ++k) {
        int ss = s - (DCV - 1) + k;
        if (ss >= 0)
            acc += w[k] * xz[(size_t)(tok - (DCV - 1) + k) * (2 * DIN) + d];
    }
    float r = acc * sigm(acc);
    xc [(size_t)tok * DIN + d] = r;
    xcb[(size_t)tok * DIN + d] = f2bf(r);
}

// ============================================================
// Selective scan: one thread per (batch, channel), 16-state in regs.
// Fuses dt/softplus, state recurrence, y = h·C + D*u, silu(z) gate,
// and bf16 output for the out_proj GEMM.
// ============================================================
__global__ __launch_bounds__(256)
void scan_kernel(const float* __restrict__ sp,  const float* __restrict__ xc,
                 const float* __restrict__ xz,  const float* __restrict__ A_log,
                 const float* __restrict__ dtw, const float* __restrict__ dtb,
                 const float* __restrict__ Dp,  unsigned short* __restrict__ yg)
{
    int q = blockIdx.x * 256 + threadIdx.x;
    if (q >= BSZ * DIN) return;
    int b = q / DIN, d = q - b * DIN;
    float A[DST];
    #pragma unroll
    for (int n = 0; n < DST; ++n) A[n] = -__expf(A_log[(size_t)d * DST + n]);
    float w = dtw[d], bb = dtb[d], Dv = Dp[d];
    float h[DST];
    #pragma unroll
    for (int n = 0; n < DST; ++n) h[n] = 0.f;
    for (int s = 0; s < SEQ; ++s) {
        size_t base = (size_t)(b * SEQ + s);
        const float* spr = sp + base * (2 * DST + 1);
        float dt = spr[0] * w + bb;
        dt = (dt > 20.f) ? dt : log1pf(__expf(dt));          // softplus
        float u  = xc[base * DIN + d];
        float du = dt * u;
        float y  = 0.f;
        #pragma unroll
        for (int n = 0; n < DST; ++n) {
            h[n] = __expf(dt * A[n]) * h[n] + du * spr[1 + n];
            y   += h[n] * spr[1 + DST + n];
        }
        y += Dv * u;
        float z  = xz[base * (2 * DIN) + DIN + d];
        yg[base * DIN + d] = f2bf(y * (z * sigm(z)));
    }
}

// ============================================================
// Orchestration
// ============================================================
extern "C" void kernel_launch(void* const* d_in, const int* in_sizes, int n_in,
                              void* d_out, int out_size, void* d_ws, size_t ws_size,
                              hipStream_t stream)
{
    const int*   token_ids  = (const int*)  d_in[0];
    const float* tok_emb    = (const float*)d_in[1];
    const float* big_emb    = (const float*)d_in[2];
    const float* big_proj   = (const float*)d_in[3];
    const float* big_scale  = (const float*)d_in[4];
    const float* skip_w     = (const float*)d_in[5];
    const float* smear_gate = (const float*)d_in[6];
    const float* in_proj_w  = (const float*)d_in[7];
    const float* conv_w     = (const float*)d_in[8];
    const float* conv_b     = (const float*)d_in[9];
    const float* x_proj_w   = (const float*)d_in[10];
    const float* dt_w       = (const float*)d_in[11];
    const float* dt_b       = (const float*)d_in[12];
    const float* A_log      = (const float*)d_in[13];
    const float* D_ssm      = (const float*)d_in[14];
    const float* out_proj_w = (const float*)d_in[15];
    const float* attn_scale = (const float*)d_in[16];
    const float* mlp_scale  = (const float*)d_in[17];
    const float* fc_w       = (const float*)d_in[18];
    const float* projm_w    = (const float*)d_in[19];
    const float* lm_head_w  = (const float*)d_in[20];

    char* ws = (char*)d_ws;
    size_t off = 0;
    auto alloc = [&](size_t bytes) -> char* {
        char* p = ws + off;
        off += (bytes + 255) & ~(size_t)255;
        return p;
    };

    float* X[NL + 1];
    for (int k = 0; k <= NL; ++k) X[k] = (float*)alloc((size_t)TOK * DIM * 4);
    float*          XS  = (float*)         alloc((size_t)TOK * DIM * 4);
    unsigned short* XN  = (unsigned short*)alloc((size_t)TOK * DIM * 2);
    float*          XZ  = (float*)         alloc((size_t)TOK * 2 * DIN * 4);
    float*          XC  = (float*)         alloc((size_t)TOK * DIN * 4);
    unsigned short* XCB = (unsigned short*)alloc((size_t)TOK * DIN * 2);
    float*          SP  = (float*)         alloc((size_t)TOK * (2 * DST + 1) * 4);
    unsigned short* YG  = (unsigned short*)alloc((size_t)TOK * DIN * 2);
    unsigned short* HB  = (unsigned short*)alloc((size_t)TOK * HID * 2);

    embed_kernel<<<TOK, 256, 0, stream>>>(token_ids, tok_emb, big_emb, big_proj,
                                          big_scale, X[0]);

    for (int i = 0; i < NL; ++i) {
        const float* skip = nullptr;
        const float* skw  = nullptr;
        if (i >= NSKIP) {
            int j = i - NSKIP;               // skips.pop(): block outputs X[6],X[5],...,X[1]
            skip = X[NSKIP - j];
            skw  = skip_w + (size_t)j * DIM;
        }
        // skip-add + smear + rmsnorm
        smear_norm_kernel<<<TOK, 256, 0, stream>>>(
            X[i], skip, skw, smear_gate + (size_t)i * DIM, XS, XN, 1);
        // xz = xn @ in_proj^T
        gemm_bf16_wmma<0><<<dim3((2 * DIN) / 128, TOK / 128), 256, 0, stream>>>(
            XN, in_proj_w + (size_t)i * 2 * DIN * DIM, XZ, nullptr, nullptr, nullptr,
            TOK, 2 * DIN, DIM);
        // xc = silu(causal dwconv(xi) + b)
        conv_kernel<<<(TOK * DIN) / 256, 256, 0, stream>>>(
            XZ, conv_w + (size_t)i * DIN * DCV, conv_b + (size_t)i * DIN, XC, XCB);
        // sp = xc @ x_proj^T  (N = 33)
        gemm_bf16_wmma<0><<<dim3(1, TOK / 128), 256, 0, stream>>>(
            XCB, x_proj_w + (size_t)i * (2 * DST + 1) * DIN, SP, nullptr, nullptr, nullptr,
            TOK, 2 * DST + 1, DIN);
        // selective scan + silu(z) gate -> bf16
        scan_kernel<<<(BSZ * DIN + 255) / 256, 256, 0, stream>>>(
            SP, XC, XZ, A_log + (size_t)i * DIN * DST,
            dt_w + (size_t)i * DIN, dt_b + (size_t)i * DIN, D_ssm + (size_t)i * DIN, YG);
        // x_mid = xs + attn_scale * (yg @ out_proj^T)
        gemm_bf16_wmma<2><<<dim3(DIM / 128, TOK / 128), 256, 0, stream>>>(
            YG, out_proj_w + (size_t)i * DIM * DIN, X[i + 1], nullptr,
            XS, attn_scale + (size_t)i * DIM, TOK, DIM, DIN);
        // xn2 = rmsnorm(x_mid)
        smear_norm_kernel<<<TOK, 256, 0, stream>>>(
            X[i + 1], nullptr, nullptr, nullptr, nullptr, XN, 0);
        // h = relu(xn2 @ fc^T)^2  (stored bf16)
        gemm_bf16_wmma<1><<<dim3(HID / 128, TOK / 128), 256, 0, stream>>>(
            XN, fc_w + (size_t)i * HID * DIM, nullptr, HB, nullptr, nullptr,
            TOK, HID, DIM);
        // x = x_mid + mlp_scale * (h @ projm^T)   (in-place residual)
        gemm_bf16_wmma<2><<<dim3(DIM / 128, TOK / 128), 256, 0, stream>>>(
            HB, projm_w + (size_t)i * DIM * HID, X[i + 1], nullptr,
            X[i + 1], mlp_scale + (size_t)i * DIM, TOK, DIM, HID);
    }

    // final rmsnorm + LM head with softcap
    smear_norm_kernel<<<TOK, 256, 0, stream>>>(
        X[NL], nullptr, nullptr, nullptr, nullptr, XN, 0);
    gemm_bf16_wmma<3><<<dim3(V_SZ / 128, TOK / 128), 256, 0, stream>>>(
        XN, lm_head_w, (float*)d_out, nullptr, nullptr, nullptr,
        TOK, V_SZ, DIM);
}
